// VectorQuantizer_29832842838409
// MI455X (gfx1250) — matrix operand order repaired and checked by reading, not compiled, not run on hardware
//
#include <hip/hip_runtime.h>

typedef __attribute__((ext_vector_type(16))) _Float16 v16h;
typedef __attribute__((ext_vector_type(8)))  float    v8f;

#define D_DIM   256
#define K_CODES 1024
#define M_BLOCK 128           // rows of z per workgroup (8 waves x 16 rows)
#define CHUNK_CODES 32        // codes staged in LDS per chunk
#define NCHUNKS (K_CODES / CHUNK_CODES)         // 32
#define NTILES  (CHUNK_CODES / 16)              // 2 N-tiles per chunk
#define KSTEPS  (D_DIM / 32)                    // 8 WMMA k-steps over D

// ---- CDNA5 async global->LDS (ASYNCcnt path, CDNA5 ISA) ----
// LDS operand: per ISA 10.2 the flat shared aperture maps addr[31:0] directly
// to the LDS byte offset, so the low 32 bits of a generic __shared__ pointer
// are the VDST LDS address.
__device__ __forceinline__ void async_ld16(unsigned lds_off, const void* gaddr) {
  asm volatile("global_load_async_to_lds_b128 %0, %1, off"
               :: "v"(lds_off), "v"((unsigned long long)(uintptr_t)gaddr)
               : "memory");
}
__device__ __forceinline__ void wait_async0() {
  asm volatile("s_wait_asynccnt 0" ::: "memory");
}

// -------- prep: split codebook into f16 hi/lo, compute ||c||^2 --------
__global__ __launch_bounds__(256) void vq_prep(const float* __restrict__ cb,
                                               float* __restrict__ cnorm,
                                               _Float16* __restrict__ cbh,
                                               _Float16* __restrict__ cbl) {
  __shared__ float red[256];
  const int k = blockIdx.x, d = threadIdx.x;
  const float c = cb[k * D_DIM + d];
  const _Float16 hi = (_Float16)c;
  const _Float16 lo = (_Float16)(c - (float)hi);
  cbh[k * D_DIM + d] = hi;
  cbl[k * D_DIM + d] = lo;
  red[d] = c * c;
  __syncthreads();
  for (int s = 128; s > 0; s >>= 1) {
    if (d < s) red[d] += red[d + s];
    __syncthreads();
  }
  if (d == 0) cnorm[k] = red[0];
}

// -------- main: async double-buffered WMMA distance GEMM + argmin --------
union SmemU {
  // [buffer][hi/lo][frag(=t*8+kk)][lane][16 f16]: 2 x 2 x 16KB = 64KB total
  struct { __align__(64) _Float16 b[2][2][NTILES * KSTEPS * 32 * 16]; } g;
  struct { int idx[M_BLOCK]; float red[256]; } f;
};

__global__ __launch_bounds__(256) void vq_main(const float*    __restrict__ z,
                                               const _Float16* __restrict__ cbh,
                                               const _Float16* __restrict__ cbl,
                                               const float*    __restrict__ cnorm,
                                               const float*    __restrict__ cbf,
                                               float*          __restrict__ out_zq,
                                               int*            __restrict__ out_idx,
                                               double*         __restrict__ partials) {
  __shared__ SmemU sm;

  const int tid  = threadIdx.x;
  const int wave = tid >> 5;
  const int lane = tid & 31;
  const int l    = lane & 15;
  const int h    = (lane >> 4) & 1;
  const int blockRow = blockIdx.x * M_BLOCK;

  // Build A fragments in registers: wave handles rows [blockRow+wave*16, +16).
  // ISA A layout (16-bit 16x32): M = lane&15 for all lanes;
  //   element e=2p+q -> K = kk*32 + (p<4 ? 2p : 2p+8) + 8*(lane>>4) + q
  const int rowA = blockRow + wave * 16 + l;
  const float2* z2 = reinterpret_cast<const float2*>(z + (size_t)rowA * D_DIM);
  v16h ahi[KSTEPS], alo[KSTEPS];
  #pragma unroll
  for (int kk = 0; kk < KSTEPS; ++kk) {
    #pragma unroll
    for (int p = 0; p < 8; ++p) {
      const int col2 = kk * 16 + p + ((p >= 4) ? 4 : 0) + 4 * h;  // (K offset)/2
      const float2 v = z2[col2];
      const _Float16 hx = (_Float16)v.x;
      const _Float16 hy = (_Float16)v.y;
      ahi[kk][2 * p]     = hx;
      ahi[kk][2 * p + 1] = hy;
      alo[kk][2 * p]     = (_Float16)(v.x - (float)hx);
      alo[kk][2 * p + 1] = (_Float16)(v.y - (float)hy);
    }
  }

  float best[8];
  int   bidx[8];
  #pragma unroll
  for (int p = 0; p < 8; ++p) { best[p] = 3.4e38f; bidx[p] = 0; }

  // piece p covers cb[code][w*8 .. w*8+7]; w*8 = kk*32 + s*16 + hh*8
  // dst slot: pre-permuted WMMA B-fragment order (32B/lane, conflict-free)
  const uint4* srcH = reinterpret_cast<const uint4*>(cbh);
  const uint4* srcL = reinterpret_cast<const uint4*>(cbl);
  unsigned slotOff[4];   // byte offset of this thread's 4 pieces inside one matrix buffer
  #pragma unroll
  for (int j = 0; j < 4; ++j) {
    const int p    = j * 256 + tid;          // 1024 pieces of 16B per matrix
    const int code = p >> 5, w = p & 31;
    const int kk = w >> 2, hh = w & 1, s = (w >> 1) & 1;
    const int t = code >> 4, ll = code & 15;
    slotOff[j] = (unsigned)((((t * 8 + kk) * 32 + hh * 16 + ll) * 2 + s) * 16);
  }

  // prologue: async-fill chunk 0 into buffer 0
  {
    const unsigned baseH = (unsigned)(uintptr_t)&sm.g.b[0][0][0];
    const unsigned baseL = (unsigned)(uintptr_t)&sm.g.b[0][1][0];
    #pragma unroll
    for (int j = 0; j < 4; ++j) {
      const int p = j * 256 + tid;
      async_ld16(baseH + slotOff[j], &srcH[p]);
      async_ld16(baseL + slotOff[j], &srcL[p]);
    }
  }

  for (int chunk = 0; chunk < NCHUNKS; ++chunk) {
    const int cur = chunk & 1;
    wait_async0();        // this wave's 8 async pieces for buffer `cur` have landed
    __syncthreads();      // every wave's pieces have landed

    if (chunk + 1 < NCHUNKS) {   // overlap next chunk's fill with this chunk's WMMAs
      const unsigned baseH = (unsigned)(uintptr_t)&sm.g.b[1 - cur][0][0];
      const unsigned baseL = (unsigned)(uintptr_t)&sm.g.b[1 - cur][1][0];
      #pragma unroll
      for (int j = 0; j < 4; ++j) {
        const int p = j * 256 + tid;
        async_ld16(baseH + slotOff[j], &srcH[(chunk + 1) * 1024 + p]);
        async_ld16(baseL + slotOff[j], &srcL[(chunk + 1) * 1024 + p]);
      }
    }

    const int codeBase = chunk * CHUNK_CODES;
    #pragma unroll
    for (int t = 0; t < NTILES; ++t) {
      v8f acc = {0.f, 0.f, 0.f, 0.f, 0.f, 0.f, 0.f, 0.f};
      #pragma unroll
      for (int kk = 0; kk < KSTEPS; ++kk) {
        const int frag = t * 8 + kk;
        const v16h bh = *reinterpret_cast<const v16h*>(&sm.g.b[cur][0][(frag * 32 + lane) * 16]);
        const v16h bl = *reinterpret_cast<const v16h*>(&sm.g.b[cur][1][(frag * 32 + lane) * 16]);
        // split-precision fp32-faithful dot: hi*hi + hi*lo + lo*hi, fp32 accumulate
        acc = __builtin_amdgcn_wmma_f32_16x16x32_f16(false, ahi[kk], false, bh, (short)0, acc, false, false);
        acc = __builtin_amdgcn_wmma_f32_16x16x32_f16(false, ahi[kk], false, bl, (short)0, acc, false, false);
        acc = __builtin_amdgcn_wmma_f32_16x16x32_f16(false, alo[kk], false, bh, (short)0, acc, false, false);
      }
      // C layout: VGPR p -> M = p + 8h, N = lane&15
      const int n  = codeBase + t * 16 + l;
      const float cn = cnorm[n];                 // L0/L2-resident, reused block-wide
      #pragma unroll
      for (int p = 0; p < 8; ++p) {
        const float dist = cn - 2.0f * acc[p];
        if (dist < best[p]) { best[p] = dist; bidx[p] = n; }  // strict < keeps first occurrence
      }
    }
  }

  // cross-lane argmin over the 16 lanes of each half-wave (masks < 16 keep halves separate)
  #pragma unroll
  for (int p = 0; p < 8; ++p) {
    float v = best[p];
    int   ii = bidx[p];
    #pragma unroll
    for (int m = 1; m < 16; m <<= 1) {
      const float ov = __shfl_xor(v, m, 32);
      const int   oi = __shfl_xor(ii, m, 32);
      if (ov < v || (ov == v && oi < ii)) { v = ov; ii = oi; }
    }
    bidx[p] = ii;
  }

  __syncthreads();  // done reading sm.g; reuse the union as sm.f
  #pragma unroll
  for (int p = 0; p < 8; ++p)
    if (l == p) sm.f.idx[wave * 16 + p + 8 * h] = bidx[p];
  __syncthreads();

  if (tid < M_BLOCK) out_idx[blockRow + tid] = sm.f.idx[tid];

  // gather z_q rows (fp32 codebook), write z_q_st (== z_q numerically), loss partial
  float lsum = 0.f;
  for (int r = 0; r < M_BLOCK; ++r) {
    const int k  = sm.f.idx[r];
    const float cq = cbf[(size_t)k * D_DIM + tid];
    const float zc = z[(size_t)(blockRow + r) * D_DIM + tid];
    out_zq[(size_t)(blockRow + r) * D_DIM + tid] = cq;
    const float d = cq - zc;
    lsum += d * d;
  }
  sm.f.red[tid] = lsum;
  __syncthreads();
  for (int s = 128; s > 0; s >>= 1) {
    if (tid < s) sm.f.red[tid] += sm.f.red[tid + s];
    __syncthreads();
  }
  if (tid == 0) partials[blockIdx.x] = (double)sm.f.red[0];
}

// -------- finalize: deterministic serial sum + warmup beta + loss scalar --------
__global__ void vq_finalize(const double* __restrict__ partials, int nb,
                            const int* __restrict__ step, float* __restrict__ out0,
                            double inv_nd) {
  if (threadIdx.x == 0 && blockIdx.x == 0) {
    double s = 0.0;
    for (int i = 0; i < nb; ++i) s += partials[i];
    const int st = *step;
    double beta = 1.0;
    if (st < 1000) {
      const double r = (double)st / 1000.0;
      beta = 0.01 + (1.0 - 0.01) * r * r;
    }
    out0[0] = (float)((1.0 + beta) * s * inv_nd);
  }
}

extern "C" void kernel_launch(void* const* d_in, const int* in_sizes, int n_in,
                              void* d_out, int out_size, void* d_ws, size_t ws_size,
                              hipStream_t stream) {
  const float* z    = (const float*)d_in[0];
  const float* cb   = (const float*)d_in[1];
  const int*   step = (const int*)d_in[2];
  const int N  = in_sizes[0] / D_DIM;   // 262144
  const int nb = N / M_BLOCK;           // 2048

  // ws layout: [0,4K) cnorm f32[1024] | [4K,20K) double partials[<=2048]
  //            [20K, +512K) cb_hi f16 | next 512K cb_lo f16
  char* ws = (char*)d_ws;
  float*    cnorm    = (float*)ws;
  double*   partials = (double*)(ws + 4096);
  _Float16* cbh      = (_Float16*)(ws + 20480);
  _Float16* cbl      = cbh + (size_t)K_CODES * D_DIM;

  float* outf   = (float*)d_out;
  float* out_zq = outf + 1;
  int*   out_idx = (int*)(outf + 1 + (size_t)N * D_DIM);

  vq_prep<<<K_CODES, 256, 0, stream>>>(cb, cnorm, cbh, cbl);
  vq_main<<<nb, 256, 0, stream>>>(z, cbh, cbl, cnorm, cb, out_zq, out_idx, partials);
  vq_finalize<<<1, 1, 0, stream>>>(partials, nb, step, outf,
                                   1.0 / ((double)N * (double)D_DIM));
}
